// GNODecoder_36112085024917
// MI455X (gfx1250) — matrix operand order repaired and checked by reading, not compiled, not run on hardware
//
#include <hip/hip_runtime.h>
#include <hip/hip_bf16.h>
#include <stdint.h>

typedef __attribute__((ext_vector_type(16))) _Float16 v16h;
typedef __attribute__((ext_vector_type(2)))  _Float16 v2h;
typedef __attribute__((ext_vector_type(8)))  float    v8f;
typedef __attribute__((ext_vector_type(2)))  float    v2f;

#define NQ_   100000
#define NL_   16384
#define NE_   1000000
#define CIN_  128
#define COUT_ 4
#define H_    64
#define PH_   256

// ---- workspace byte offsets ----
#define WS_SUMS 0                                   // NQ*128 f32 = 51,200,000 B
#define WS_CNT  (NQ_*CIN_*4)                        // NQ f32    =    400,000 B
#define WS_W0P  (WS_CNT + NQ_*4)                    //  2048 f16 (KT=1,NT=4)
#define WS_W1P  (WS_W0P + 2048*2)                   //  4096 f16 (KT=2,NT=4)
#define WS_W2P  (WS_W1P + 4096*2)                   //  8192 f16 (KT=2,NT=8)
#define WS_P0P  (WS_W2P + 8192*2)                   // 32768 f16 (KT=4,NT=16)
#define WS_P1P  (WS_P0P + 32768*2)                  //  4096 f16 (KT=8,NT=1)

__device__ __forceinline__ float fast_exp2(float x) {
#if __has_builtin(__builtin_amdgcn_exp2f)
    return __builtin_amdgcn_exp2f(x);
#else
    return exp2f(x);
#endif
}
__device__ __forceinline__ float fast_rcp(float x) {
#if __has_builtin(__builtin_amdgcn_rcpf)
    return __builtin_amdgcn_rcpf(x);
#else
    return 1.0f / x;
#endif
}

// Branch-free tanh-gelu: gelu(x) = x * sigmoid(1.5957691*x*(1+0.044715*x^2))
//                               = x / (1 + exp2(-2.3022059*x*(1+0.044715*x^2)))
// (log2(e) folded into the constant; saturates correctly at +/-inf, no branches,
//  one v_exp_f32 + one v_rcp_f32 — both TRANS, co-execute with WMMA)
__device__ __forceinline__ float gelu_f(float x) {
    float x2 = x * x;
    float z  = -2.302205865f * x * fmaf(0.044715f, x2, 1.0f);
    float e  = fast_exp2(z);
    return x * fast_rcp(1.0f + e);
}

// A-fragment K mapping for 16-bit A 16x32 (contiguous, matches B packing):
// lanes 0-15 : K=t
// lanes 16-31: K=t+16
__device__ __forceinline__ int a_kmap(int t, int half) {
    return t + 16 * half;
}

__device__ __forceinline__ v8f wmma_f16(v16h a, v16h b, v8f c) {
    return __builtin_amdgcn_wmma_f32_16x16x32_f16(false, a, false, b, (short)0, c,
                                                  false, false);
}

// ---------------- kernel 0a: zero sums + cnt ----------------
__global__ void gno_zero_kernel(float* __restrict__ sums, float* __restrict__ cnt) {
    long long i = (long long)blockIdx.x * blockDim.x + threadIdx.x;
    long long n = (long long)NQ_ * CIN_;
    if (i < n)            sums[i] = 0.0f;
    else if (i < n + NQ_) cnt[i - n] = 0.0f;
}

// ---------------- kernel 0b: pack row-major f32 weights into f16 B-fragment order
// B frag (kt,nt): lane holds col N = nt*16 + lane%16; element t holds
// K = kt*32 + (lane/16)*16 + t. Packed index = ((kt*NT+nt)*32 + lane)*16 + t.
__global__ void gno_pack_kernel(const float* __restrict__ src, _Float16* __restrict__ dst,
                                int Kreal, int Nreal, int KT, int NT) {
    int id = blockIdx.x * blockDim.x + threadIdx.x;
    int total = KT * NT * 32;
    if (id >= total) return;
    int lane = id & 31;
    int fn   = id >> 5;
    int nt   = fn % NT;
    int kt   = fn / NT;
    int half = lane >> 4;
    int N    = nt * 16 + (lane & 15);
#pragma unroll
    for (int t = 0; t < 16; t++) {
        int K   = kt * 32 + half * 16 + t;
        float v = (K < Kreal && N < Nreal) ? src[K * Nreal + N] : 0.0f;
        dst[id * 16 + t] = (_Float16)v;
    }
}

// ---------------- kernel 1: per-edge MLP (WMMA) + linear transform + scatter
__global__ __launch_bounds__(256) void gno_edge_kernel(
    const float* __restrict__ rnd,   const float* __restrict__ qpos,
    const float* __restrict__ lpos,  const int* __restrict__ edst,
    const int* __restrict__ esrc,    const _Float16* __restrict__ wpack,  // W0P..W2P contiguous
    const float* __restrict__ b0,    const float* __restrict__ b1,
    const float* __restrict__ b2,    float* __restrict__ sums,
    float* __restrict__ cnt) {
    __shared__ _Float16 wlds[14336];        // 2048 + 4096 + 8192 f16 fragments
    __shared__ _Float16 hst[8][16 * 66];    // per-wave 16x64 activation staging (padded)
    __shared__ int      sIdx[8][16];
    __shared__ int      dIdx[8][16];

    const int tid = threadIdx.x;
    {   // cooperative fragment-weight preload (already f16-packed in ws)
        const uint32_t* s = (const uint32_t*)wpack;
        uint32_t*       d = (uint32_t*)wlds;
        for (int i = tid; i < 14336 / 2; i += 256) d[i] = s[i];
    }
    __syncthreads();

    const int lane = tid & 31;
    const int wave = tid >> 5;
    const int half = lane >> 4;
    const int col  = lane & 15;
    const int NTILES = NE_ / 16;

    for (int tile = blockIdx.x * 8 + wave; tile < NTILES; tile += gridDim.x * 8) {
        const int e0 = tile * 16;
        int myD = 0;
        v16h a0 = {};
        if (lane < 16) {           // A layout: K=0..5 live only in lanes 0-15 (t=K)
            myD = edst[e0 + lane];
            int myS = esrc[e0 + lane];
            sIdx[wave][lane] = myS;
            dIdx[wave][lane] = myD;
            a0[0] = (_Float16)qpos[myD * 3 + 0];
            a0[1] = (_Float16)qpos[myD * 3 + 1];
            a0[2] = (_Float16)qpos[myD * 3 + 2];
            a0[3] = (_Float16)lpos[myS * 3 + 0];
            a0[4] = (_Float16)lpos[myS * 3 + 1];
            a0[5] = (_Float16)lpos[myS * 3 + 2];
        }

        // ---- layer 0: [16,32pad] x [32,64], gelu, stage ----
        for (int n = 0; n < 4; n++) {
            v16h b = *(const v16h*)&wlds[n * 512 + lane * 16];
            v8f  c = {};
            c = wmma_f16(a0, b, c);
            float bias = b0[n * 16 + col];
#pragma unroll
            for (int j = 0; j < 8; j++) {
                int row = j + 8 * half;
                hst[wave][row * 66 + n * 16 + col] = (_Float16)gelu_f(c[j] + bias);
            }
        }

        // ---- build layer-1 A fragments from staging ----
        v16h a1[2];
#pragma unroll
        for (int kt = 0; kt < 2; kt++) {
#pragma unroll
            for (int p = 0; p < 8; p++) {
                int t = 2 * p;
                int K = kt * 32 + a_kmap(t, half);
                v2h v = *(const v2h*)&hst[wave][col * 66 + K];
                a1[kt][t] = v[0];
                a1[kt][t + 1] = v[1];
            }
        }

        // ---- layer 1: [16,64] x [64,64], gelu, restage ----
        for (int n = 0; n < 4; n++) {
            v8f  c = {};
            v16h b = *(const v16h*)&wlds[2048 + (0 * 4 + n) * 512 + lane * 16];
            c = wmma_f16(a1[0], b, c);
            b = *(const v16h*)&wlds[2048 + (1 * 4 + n) * 512 + lane * 16];
            c = wmma_f16(a1[1], b, c);
            float bias = b1[n * 16 + col];
#pragma unroll
            for (int j = 0; j < 8; j++) {
                int row = j + 8 * half;
                hst[wave][row * 66 + n * 16 + col] = (_Float16)gelu_f(c[j] + bias);
            }
        }

        // ---- build layer-2 A fragments ----
        v16h a2[2];
#pragma unroll
        for (int kt = 0; kt < 2; kt++) {
#pragma unroll
            for (int p = 0; p < 8; p++) {
                int t = 2 * p;
                int K = kt * 32 + a_kmap(t, half);
                v2h v = *(const v2h*)&hst[wave][col * 66 + K];
                a2[kt][t] = v[0];
                a2[kt][t + 1] = v[1];
            }
        }

        int sj[8], dj[8];
#pragma unroll
        for (int j = 0; j < 8; j++) {
            int row = j + 8 * half;
            sj[j] = sIdx[wave][row];
            dj[j] = dIdx[wave][row];
        }

        // ---- layer 2: [16,64] x [64,128]; fuse k * rndata[src] + atomic scatter ----
        for (int n = 0; n < 8; n++) {
            v8f  c = {};
            v16h b = *(const v16h*)&wlds[6144 + (0 * 8 + n) * 512 + lane * 16];
            c = wmma_f16(a2[0], b, c);
            b = *(const v16h*)&wlds[6144 + (1 * 8 + n) * 512 + lane * 16];
            c = wmma_f16(a2[1], b, c);
            int   ch   = n * 16 + col;
            float bias = b2[ch];
#pragma unroll
            for (int j = 0; j < 8; j++) {
                float v = (c[j] + bias) * rnd[sj[j] * CIN_ + ch];
                atomicAdd(&sums[dj[j] * CIN_ + ch], v);
            }
        }
        if (lane < 16) atomicAdd(&cnt[myD], 1.0f);
    }
}

// ---------------- kernel 2: mean + projection MLP (WMMA) ----------------
__global__ __launch_bounds__(128) void gno_proj_kernel(
    const float* __restrict__ sums, const float* __restrict__ cnt,
    const _Float16* __restrict__ p0p, const _Float16* __restrict__ p1p,
    const float* __restrict__ pb0, const float* __restrict__ pb1,
    float* __restrict__ out) {
    __shared__ _Float16 hst[4][16 * 264];   // per-wave 16x256 staging (padded)

    const int tid  = threadIdx.x;
    const int lane = tid & 31;
    const int wave = tid >> 5;
    const int half = lane >> 4;
    const int col  = lane & 15;

    const int qt = blockIdx.x * 4 + wave;   // wave-uniform guard (EXEC stays all-1s)
    if (qt >= NQ_ / 16) return;
    const int q0 = qt * 16;
    const int q  = q0 + col;

    float cv   = cnt[q];
    float cinv = fast_rcp(cv < 1.0f ? 1.0f : cv);

    // A fragments for agg = sums/cnt : [16 q, 128 ch] -> 4 K-fragments
    v16h a[4];
#pragma unroll
    for (int kt = 0; kt < 4; kt++) {
#pragma unroll
        for (int p = 0; p < 8; p++) {
            int t = 2 * p;
            int K = kt * 32 + a_kmap(t, half);
            v2f v = *(const v2f*)&sums[(long long)q * CIN_ + K];
            a[kt][t]     = (_Float16)(v[0] * cinv);
            a[kt][t + 1] = (_Float16)(v[1] * cinv);
        }
    }

    // ---- layer P0: [16,128] x [128,256], gelu, stage ----
    for (int nt = 0; nt < 16; nt++) {
        v8f c = {};
#pragma unroll
        for (int kt = 0; kt < 4; kt++) {
            v16h b = *(const v16h*)&p0p[((kt * 16 + nt) * 32 + lane) * 16];
            c = wmma_f16(a[kt], b, c);
        }
        float bias = pb0[nt * 16 + col];
#pragma unroll
        for (int j = 0; j < 8; j++) {
            int row = j + 8 * half;
            hst[wave][row * 264 + nt * 16 + col] = (_Float16)gelu_f(c[j] + bias);
        }
    }

    // ---- layer P1: [16,256] x [256,16pad] ----
    v8f c = {};
    for (int kt = 0; kt < 8; kt++) {
        v16h a2;
#pragma unroll
        for (int p = 0; p < 8; p++) {
            int t = 2 * p;
            int K = kt * 32 + a_kmap(t, half);
            v2h v = *(const v2h*)&hst[wave][col * 264 + K];
            a2[t] = v[0];
            a2[t + 1] = v[1];
        }
        v16h b = *(const v16h*)&p1p[(kt * 32 + lane) * 16];
        c = wmma_f16(a2, b, c);
    }

    if (col < COUT_) {
        float bias = pb1[col];
#pragma unroll
        for (int j = 0; j < 8; j++) {
            int row = j + 8 * half;
            out[(long long)(q0 + row) * COUT_ + col] = c[j] + bias;
        }
    }
}

extern "C" void kernel_launch(void* const* d_in, const int* in_sizes, int n_in,
                              void* d_out, int out_size, void* d_ws, size_t ws_size,
                              hipStream_t stream) {
    const float* rnd  = (const float*)d_in[0];
    const float* qpos = (const float*)d_in[1];
    const float* lpos = (const float*)d_in[2];
    const int*   edst = (const int*)d_in[3];
    const int*   esrc = (const int*)d_in[4];
    const float* W0 = (const float*)d_in[5];
    const float* b0 = (const float*)d_in[6];
    const float* W1 = (const float*)d_in[7];
    const float* b1 = (const float*)d_in[8];
    const float* W2 = (const float*)d_in[9];
    const float* b2 = (const float*)d_in[10];
    const float* P0 = (const float*)d_in[11];
    const float* pb0 = (const float*)d_in[12];
    const float* P1 = (const float*)d_in[13];
    const float* pb1 = (const float*)d_in[14];
    float* out = (float*)d_out;

    char* ws = (char*)d_ws;
    float*     sums = (float*)(ws + WS_SUMS);
    float*     cnt  = (float*)(ws + WS_CNT);
    _Float16*  w0p  = (_Float16*)(ws + WS_W0P);
    _Float16*  w1p  = (_Float16*)(ws + WS_W1P);
    _Float16*  w2p  = (_Float16*)(ws + WS_W2P);
    _Float16*  p0p  = (_Float16*)(ws + WS_P0P);
    _Float16*  p1p  = (_Float16*)(ws + WS_P1P);

    // 0a: zero accumulators
    {
        long long n = (long long)NQ_ * (CIN_ + 1);
        int grid = (int)((n + 255) / 256);
        gno_zero_kernel<<<grid, 256, 0, stream>>>(sums, cnt);
    }
    // 0b: pack weights into f16 B-fragment order
    gno_pack_kernel<<<1, 128, 0, stream>>>(W0, w0p, 6, 64, 1, 4);
    gno_pack_kernel<<<1, 256, 0, stream>>>(W1, w1p, 64, 64, 2, 4);
    gno_pack_kernel<<<2, 256, 0, stream>>>(W2, w2p, 64, 128, 2, 8);
    gno_pack_kernel<<<8, 256, 0, stream>>>(P0, p0p, 128, 256, 4, 16);
    gno_pack_kernel<<<1, 256, 0, stream>>>(P1, p1p, 256, 4, 8, 1);

    // 1: edge MLP + scatter (grid-stride; 8 wave-tiles of 16 edges per block)
    gno_edge_kernel<<<4096, 256, 0, stream>>>(rnd, qpos, lpos, edst, esrc, w0p,
                                              b0, b1, b2, sums, cnt);

    // 2: mean + projection (4 wave-tiles of 16 queries per block)
    {
        int grid = (NQ_ / 16 + 3) / 4;   // 6250 wave-tiles
        gno_proj_kernel<<<grid, 128, 0, stream>>>(sums, cnt, p0p, p1p, pb0, pb1, out);
    }
}
